// HybridTransformerBlock_65481071396852
// MI455X (gfx1250) — matrix operand order repaired and checked by reading, not compiled, not run on hardware
//
#include <hip/hip_runtime.h>
#include <hip/hip_bf16.h>

#define DEV __device__ __forceinline__
#define LDS_AS   __attribute__((address_space(3)))
#define GLOB_AS  __attribute__((address_space(1)))

#if __has_builtin(__builtin_amdgcn_global_load_async_to_lds_b128) && \
    __has_builtin(__builtin_amdgcn_s_wait_asynccnt)
#define USE_ASYNC_LDS 1
#else
#define USE_ASYNC_LDS 0
#endif

typedef __attribute__((ext_vector_type(4)))  int          v4i;
typedef __attribute__((ext_vector_type(8)))  float        v8f;
typedef __attribute__((ext_vector_type(8)))  unsigned int v8u;
typedef __attribute__((ext_vector_type(16))) __bf16       v16bf;

static_assert(sizeof(v8u) == 32, "frag size");
static_assert(sizeof(v16bf) == 32, "frag size");

// ---- model dimensions (compile-time) ----
constexpr int E_  = 1024;
constexpr int H_  = 16;
constexpr int F_  = 4096;
constexpr int B_  = 4;
constexpr int S_  = 2048;
constexpr int HD_ = 64;            // head dim
constexpr int BS_ = B_ * S_;       // 8192 token rows

// ---------- helpers ----------
DEV unsigned short f32_to_bf16(float f) {
  unsigned int u = __builtin_bit_cast(unsigned int, f);
  u += 0x7FFFu + ((u >> 16) & 1u);               // round-to-nearest-even
  return (unsigned short)(u >> 16);
}

DEV v8f wmma_bf16(v8u a, v8u b, v8f c) {
  // D(16x16,f32) = A(16x32,bf16) x B(32x16,bf16) + C
  return __builtin_amdgcn_wmma_f32_16x16x32_bf16(
      false, __builtin_bit_cast(v16bf, a),
      false, __builtin_bit_cast(v16bf, b),
      (short)0, c, false, false);
}

// A fragment 16x32 from a row-major [row][k] tile (ISA 7.12.2 16-bit A layout):
// lanes 0-15: row=lane, K in {0..7,16..23}; lanes 16-31: same rows, K {8..15,24..31}
DEV v8u frag_a_rowmajor(const unsigned short* s, int stride, int kk) {
  const int lane = threadIdx.x & 31;
  const int row  = lane & 15;
  const int hi   = lane >> 4;
  v8u f;
#pragma unroll
  for (int v = 0; v < 8; ++v) {
    const int k = kk + 2 * v + 8 * ((v >= 4 ? 1 : 0) + hi);
    f[v] = *reinterpret_cast<const unsigned int*>(s + row * stride + k);
  }
  return f;
}

// B fragment 32x16 where B[k][n] = tile[n][k] (NT: tile stored [n][k] row-major).
// lanes 0-15 hold K {0..7,16..23} of column n=lane; lanes 16-31 hold K {8..15,24..31}.
DEV v8u frag_b_nt(const unsigned short* s, int stride, int kk) {
  const int lane = threadIdx.x & 31;
  const int n    = lane & 15;
  const int kb   = (lane >> 4) * 8;
  v8u f;
#pragma unroll
  for (int v = 0; v < 8; ++v)
    f[v] = *reinterpret_cast<const unsigned int*>(s + n * stride + kk + kb + 2 * v + (v >= 4 ? 8 : 0));
  return f;
}

// async (or fallback sync) 16-byte global -> LDS copy, per lane
DEV void copy16_g2l(const unsigned short* g, unsigned short* l) {
#if USE_ASYNC_LDS
  unsigned short* gnc = const_cast<unsigned short*>(g);
  __builtin_amdgcn_global_load_async_to_lds_b128(
      (GLOB_AS v4i*)gnc, (LDS_AS v4i*)l, 0, 0);
#else
  *reinterpret_cast<uint4*>(l) = *reinterpret_cast<const uint4*>(g);
#endif
}

DEV void async_join() {
#if USE_ASYNC_LDS
  __builtin_amdgcn_s_wait_asynccnt(0);
#endif
  __syncthreads();
}

// ---------- fp32 -> bf16 cast ----------
__global__ __launch_bounds__(256) void cast_f32_bf16_kernel(
    const float* __restrict__ in, unsigned short* __restrict__ out, int n4) {
  const int i = blockIdx.x * blockDim.x + threadIdx.x;
  if (i >= n4) return;
  const float4 v = reinterpret_cast<const float4*>(in)[i];
  const unsigned int w0 = (unsigned int)f32_to_bf16(v.x) | ((unsigned int)f32_to_bf16(v.y) << 16);
  const unsigned int w1 = (unsigned int)f32_to_bf16(v.z) | ((unsigned int)f32_to_bf16(v.w) << 16);
  reinterpret_cast<uint2*>(out)[i] = make_uint2(w0, w1);
}

// ---------- bf16 NT GEMM: C[M,N] = A[M,K] * B[N,K]^T (+bias)(+relu) ----------
// Double-buffered LDS pipeline: stage tile k+1 via ASYNC-to-LDS while WMMAs
// consume tile k; s_wait_asynccnt 0 + barrier gates each stage.
template <bool OUT_BF16, bool RELU, bool HAS_BIAS>
__global__ __launch_bounds__(256) void gemm_nt_kernel(
    const unsigned short* __restrict__ A, const unsigned short* __restrict__ Bw,
    const float* __restrict__ bias, void* __restrict__ Cv, int M, int N, int K) {
  constexpr int BM = 128, BN = 128, BK = 64, STR = BK + 8;   // 144B row stride: 16B aligned
  __shared__ unsigned short As[2][BM * STR];
  __shared__ unsigned short Bs[2][BN * STR];

  const int tid  = threadIdx.x;
  const int wave = tid >> 5;
  const int wm   = wave & 3;          // 4 waves along M
  const int wn   = wave >> 2;         // 2 waves along N
  const int m0   = blockIdx.y * BM;
  const int n0   = blockIdx.x * BN;

  const v8f vzero = {0.f, 0.f, 0.f, 0.f, 0.f, 0.f, 0.f, 0.f};
  v8f acc[2][4];
#pragma unroll
  for (int i = 0; i < 2; ++i)
#pragma unroll
    for (int j = 0; j < 4; ++j) acc[i][j] = vzero;

  auto stage = [&](int buf, int k0) {
#pragma unroll
    for (int i = 0; i < 4; ++i) {                 // 256 thr * 4 * 16B = 16KB per matrix
      const int idx = tid + i * 256;
      const int row = idx >> 3;
      const int c8  = (idx & 7) * 8;
      copy16_g2l(&A [(size_t)(m0 + row) * K + k0 + c8], &As[buf][row * STR + c8]);
      copy16_g2l(&Bw[(size_t)(n0 + row) * K + k0 + c8], &Bs[buf][row * STR + c8]);
    }
  };

  stage(0, 0);
  for (int k0 = 0; k0 < K; k0 += BK) {
    const int cur = (k0 / BK) & 1;
    async_join();                                  // cur buffer resident in LDS
    if (k0 + BK < K) stage(cur ^ 1, k0 + BK);      // prefetch next tile into other buffer
#pragma unroll
    for (int kk = 0; kk < BK; kk += 32) {
      v8u af[2], bf[4];
#pragma unroll
      for (int mf = 0; mf < 2; ++mf)
        af[mf] = frag_a_rowmajor(&As[cur][(wm * 32 + mf * 16) * STR], STR, kk);
#pragma unroll
      for (int nf = 0; nf < 4; ++nf)
        bf[nf] = frag_b_nt(&Bs[cur][(wn * 64 + nf * 16) * STR], STR, kk);
#pragma unroll
      for (int mf = 0; mf < 2; ++mf)
#pragma unroll
        for (int nf = 0; nf < 4; ++nf)
          acc[mf][nf] = wmma_bf16(af[mf], bf[nf], acc[mf][nf]);
    }
  }

  // epilogue: C layout (ISA 7.12.2): VGPR r -> row r (lanes 0-15) / r+8 (lanes 16-31)
  const int lane  = tid & 31;
  const int col16 = lane & 15;
  const int rhi   = (lane >> 4) * 8;
#pragma unroll
  for (int nf = 0; nf < 4; ++nf) {
    const int col  = n0 + wn * 64 + nf * 16 + col16;
    const float bv = HAS_BIAS ? bias[col] : 0.0f;
#pragma unroll
    for (int mf = 0; mf < 2; ++mf) {
#pragma unroll
      for (int r = 0; r < 8; ++r) {
        const int row = m0 + wm * 32 + mf * 16 + rhi + r;
        float v = acc[mf][nf][r] + bv;
        if (RELU) v = fmaxf(v, 0.0f);
        if (OUT_BF16)
          reinterpret_cast<unsigned short*>(Cv)[(size_t)row * N + col] = f32_to_bf16(v);
        else
          reinterpret_cast<float*>(Cv)[(size_t)row * N + col] = v;
      }
    }
  }
}

// ---------- flash attention: one wave = 16 query rows, stream K/V in 32-key tiles ----------
// K staged row-major (async); V staged TRANSPOSED (Vt[d][key]) so the P*V
// B-fragments are contiguous ds_load_b32 reads instead of 16-bit gathers.
__global__ __launch_bounds__(128) void flash_attn_kernel(
    const unsigned short* __restrict__ Q, const unsigned short* __restrict__ Kb,
    const unsigned short* __restrict__ Vb, unsigned short* __restrict__ ctx) {
  constexpr int STR  = HD_ + 8;   // 72: K tile row stride
  constexpr int TSTR = 40;        // Vt row stride (32 keys + pad)
  constexpr int PSTR = 40;        // P-tile scratch stride
  __shared__ unsigned short Ks[32 * STR];
  __shared__ unsigned short Vt[HD_ * TSTR];
  __shared__ unsigned short Ps[4 * 16 * PSTR];

  const int tid  = threadIdx.x;
  const int wave = tid >> 5;
  const int lane = tid & 31;
  const int bh   = blockIdx.y;
  const int b    = bh >> 4;
  const int h    = bh & 15;
  const int q0   = blockIdx.x * 64 + wave * 16;

  // Q rows live at [b*S + s][h*64 + d] in the projection output
  const size_t qoff = ((size_t)b * S_ + q0) * E_ + (size_t)h * HD_;
  v8u qf[2];
#pragma unroll
  for (int c = 0; c < 2; ++c) qf[c] = frag_a_rowmajor(Q + qoff, E_, c * 32);

  const v8f vzero = {0.f, 0.f, 0.f, 0.f, 0.f, 0.f, 0.f, 0.f};
  v8f o[4];
#pragma unroll
  for (int f = 0; f < 4; ++f) o[f] = vzero;
  float m_run[8], l_run[8];
#pragma unroll
  for (int r = 0; r < 8; ++r) { m_run[r] = -__builtin_inff(); l_run[r] = 0.0f; }

  for (int j = 0; j < S_; j += 32) {
#pragma unroll
    for (int i = 0; i < 2; ++i) {               // cooperative K/V tile load (32x64 bf16)
      const int idx = tid + i * 128;
      const int row = idx >> 3;
      const int c8  = (idx & 7) * 8;
      const size_t src = ((size_t)b * S_ + j + row) * E_ + (size_t)h * HD_ + c8;
      copy16_g2l(&Kb[src], &Ks[row * STR + c8]);            // K: row-major (async)
      const uint4 vv = *reinterpret_cast<const uint4*>(&Vb[src]);
      const unsigned short* vs = reinterpret_cast<const unsigned short*>(&vv);
#pragma unroll
      for (int e = 0; e < 8; ++e)                           // V: transpose into Vt[d][key]
        Vt[(c8 + e) * TSTR + row] = vs[e];
    }
    async_join();

    // scores: S = Q (16x64) x K^T, split hd into two k=32 chunks; two 16-key subtiles
    v8f s0 = vzero, s1 = vzero;
#pragma unroll
    for (int c = 0; c < 2; ++c) {
      s0 = wmma_bf16(qf[c], frag_b_nt(&Ks[0],        STR, c * 32), s0);
      s1 = wmma_bf16(qf[c], frag_b_nt(&Ks[16 * STR], STR, c * 32), s1);
    }

    // streaming softmax over this 32-key tile (per-row across 16-lane halves)
#pragma unroll
    for (int r = 0; r < 8; ++r) {
      const float a0 = s0[r] * 0.125f;          // 1/sqrt(64)
      const float a1 = s1[r] * 0.125f;
      float mt = fmaxf(a0, a1);
#pragma unroll
      for (int off = 8; off >= 1; off >>= 1) mt = fmaxf(mt, __shfl_xor(mt, off, 32));
      const float mn = fmaxf(m_run[r], mt);
      const float p0 = __expf(a0 - mn);
      const float p1 = __expf(a1 - mn);
      float lt = p0 + p1;
#pragma unroll
      for (int off = 8; off >= 1; off >>= 1) lt += __shfl_xor(lt, off, 32);
      const float sc = __expf(m_run[r] - mn);
      l_run[r] = l_run[r] * sc + lt;
      m_run[r] = mn;
#pragma unroll
      for (int f = 0; f < 4; ++f) o[f][r] *= sc;
      // stash P (C-layout) into per-wave LDS so it can be re-read in A-layout
      const int rl = r + ((lane >> 4) << 3);
      const int cl = lane & 15;
      Ps[(wave * 16 + rl) * PSTR + cl]      = f32_to_bf16(p0);
      Ps[(wave * 16 + rl) * PSTR + 16 + cl] = f32_to_bf16(p1);
    }

    // O += P (16x32) x V (32x64): B[k][n] = Vt[f*16+n][k] -> contiguous NT reads
    const v8u pf = frag_a_rowmajor(&Ps[wave * 16 * PSTR], PSTR, 0);
#pragma unroll
    for (int f = 0; f < 4; ++f)
      o[f] = wmma_bf16(pf, frag_b_nt(&Vt[(f * 16) * TSTR], TSTR, 0), o[f]);
    __syncthreads();
  }

  // normalize and emit context in [b,s][h*64+d] layout (== transpose+reshape)
#pragma unroll
  for (int r = 0; r < 8; ++r) {
    const float inv = 1.0f / l_run[r];
    const int rl = r + ((lane >> 4) << 3);
    const size_t row = (size_t)b * S_ + q0 + rl;
#pragma unroll
    for (int f = 0; f < 4; ++f) {
      const int col = h * HD_ + f * 16 + (lane & 15);
      ctx[row * E_ + col] = f32_to_bf16(o[f][r] * inv);
    }
  }
}

// ---------- residual add + LayerNorm (fp32), optional bf16 mirror ----------
__global__ __launch_bounds__(256) void add_ln_kernel(
    const float* __restrict__ x, const float* __restrict__ y,
    const float* __restrict__ g, const float* __restrict__ bta,
    float* __restrict__ outf, unsigned short* __restrict__ outb) {
  __shared__ float red[2][8];
  const int row = blockIdx.x;
  const int tid = threadIdx.x;
  const size_t base = (size_t)row * E_;
  const float4 xv = reinterpret_cast<const float4*>(x + base)[tid];
  const float4 yv = reinterpret_cast<const float4*>(y + base)[tid];
  const float v0 = xv.x + yv.x, v1 = xv.y + yv.y, v2 = xv.z + yv.z, v3 = xv.w + yv.w;
  float s1 = v0 + v1 + v2 + v3;
  float s2 = v0 * v0 + v1 * v1 + v2 * v2 + v3 * v3;
#pragma unroll
  for (int off = 16; off >= 1; off >>= 1) {
    s1 += __shfl_xor(s1, off, 32);
    s2 += __shfl_xor(s2, off, 32);
  }
  if ((tid & 31) == 0) { red[0][tid >> 5] = s1; red[1][tid >> 5] = s2; }
  __syncthreads();
  float t1 = 0.0f, t2 = 0.0f;
#pragma unroll
  for (int i = 0; i < 8; ++i) { t1 += red[0][i]; t2 += red[1][i]; }
  const float mean = t1 * (1.0f / E_);
  const float var  = t2 * (1.0f / E_) - mean * mean;
  const float rstd = rsqrtf(var + 1e-5f);
  const int c0 = tid * 4;
  const float vals[4] = {v0, v1, v2, v3};
  float4 ov;
  float* op = &ov.x;
#pragma unroll
  for (int i = 0; i < 4; ++i) {
    const float o = (vals[i] - mean) * rstd * g[c0 + i] + bta[c0 + i];
    op[i] = o;
    if (outb) outb[base + c0 + i] = f32_to_bf16(o);
  }
  reinterpret_cast<float4*>(outf + base)[tid] = ov;
}

// ---------- host launcher ----------
extern "C" void kernel_launch(void* const* d_in, const int* in_sizes, int n_in,
                              void* d_out, int out_size, void* d_ws, size_t ws_size,
                              hipStream_t stream) {
  (void)in_sizes; (void)n_in; (void)out_size; (void)ws_size;
  const float* x   = (const float*)d_in[0];
  const float* Wq  = (const float*)d_in[1];
  const float* Wk  = (const float*)d_in[2];
  const float* Wv  = (const float*)d_in[3];
  const float* Wo  = (const float*)d_in[4];
  const float* g1  = (const float*)d_in[5];
  const float* b1  = (const float*)d_in[6];
  const float* g2  = (const float*)d_in[7];
  const float* b2  = (const float*)d_in[8];
  const float* W1  = (const float*)d_in[9];
  const float* bb1 = (const float*)d_in[10];
  const float* W2  = (const float*)d_in[11];
  const float* bb2 = (const float*)d_in[12];

  char* ws = (char*)d_ws;
  size_t off = 0;
  auto alloc = [&](size_t bytes) -> void* {
    void* p = ws + off;
    off = (off + bytes + 255) & ~(size_t)255;
    return p;
  };
  unsigned short* xh   = (unsigned short*)alloc((size_t)BS_ * E_ * 2);
  unsigned short* wqh  = (unsigned short*)alloc((size_t)E_ * E_ * 2);
  unsigned short* wkh  = (unsigned short*)alloc((size_t)E_ * E_ * 2);
  unsigned short* wvh  = (unsigned short*)alloc((size_t)E_ * E_ * 2);
  unsigned short* woh  = (unsigned short*)alloc((size_t)E_ * E_ * 2);
  unsigned short* w1h  = (unsigned short*)alloc((size_t)F_ * E_ * 2);
  unsigned short* w2h  = (unsigned short*)alloc((size_t)E_ * F_ * 2);
  unsigned short* qb   = (unsigned short*)alloc((size_t)BS_ * E_ * 2);
  unsigned short* kb   = (unsigned short*)alloc((size_t)BS_ * E_ * 2);
  unsigned short* vb   = (unsigned short*)alloc((size_t)BS_ * E_ * 2);
  unsigned short* ctx  = (unsigned short*)alloc((size_t)BS_ * E_ * 2);
  float*          attnf= (float*)alloc((size_t)BS_ * E_ * 4);
  float*          hf   = (float*)alloc((size_t)BS_ * E_ * 4);
  unsigned short* hb   = (unsigned short*)alloc((size_t)BS_ * E_ * 2);
  unsigned short* ffn1 = (unsigned short*)alloc((size_t)BS_ * F_ * 2);
  float*          ffn2 = (float*)alloc((size_t)BS_ * E_ * 4);

  const int TPB = 256;
  // casts to bf16
  cast_f32_bf16_kernel<<<(BS_ * E_ / 4) / TPB, TPB, 0, stream>>>(x,  xh,  BS_ * E_ / 4);
  cast_f32_bf16_kernel<<<(E_ * E_ / 4)  / TPB, TPB, 0, stream>>>(Wq, wqh, E_ * E_ / 4);
  cast_f32_bf16_kernel<<<(E_ * E_ / 4)  / TPB, TPB, 0, stream>>>(Wk, wkh, E_ * E_ / 4);
  cast_f32_bf16_kernel<<<(E_ * E_ / 4)  / TPB, TPB, 0, stream>>>(Wv, wvh, E_ * E_ / 4);
  cast_f32_bf16_kernel<<<(E_ * E_ / 4)  / TPB, TPB, 0, stream>>>(Wo, woh, E_ * E_ / 4);
  cast_f32_bf16_kernel<<<(F_ * E_ / 4)  / TPB, TPB, 0, stream>>>(W1, w1h, F_ * E_ / 4);
  cast_f32_bf16_kernel<<<(E_ * F_ / 4)  / TPB, TPB, 0, stream>>>(W2, w2h, E_ * F_ / 4);

  // QKV projections (NT, bf16 out)
  dim3 gq(E_ / 128, BS_ / 128);
  gemm_nt_kernel<true,  false, false><<<gq, 256, 0, stream>>>(xh, wqh, nullptr, qb, BS_, E_, E_);
  gemm_nt_kernel<true,  false, false><<<gq, 256, 0, stream>>>(xh, wkh, nullptr, kb, BS_, E_, E_);
  gemm_nt_kernel<true,  false, false><<<gq, 256, 0, stream>>>(xh, wvh, nullptr, vb, BS_, E_, E_);

  // flash attention
  dim3 ga(S_ / 64, B_ * H_);
  flash_attn_kernel<<<ga, 128, 0, stream>>>(qb, kb, vb, ctx);

  // output projection (f32 out) + residual/LN1
  gemm_nt_kernel<false, false, false><<<gq, 256, 0, stream>>>(ctx, woh, nullptr, attnf, BS_, E_, E_);
  add_ln_kernel<<<BS_, 256, 0, stream>>>(x, attnf, g1, b1, hf, hb);

  // FFN
  dim3 gf(F_ / 128, BS_ / 128);
  gemm_nt_kernel<true,  true,  true ><<<gf, 256, 0, stream>>>(hb,   w1h, bb1, ffn1, BS_, F_, E_);
  gemm_nt_kernel<false, false, true ><<<gq, 256, 0, stream>>>(ffn1, w2h, bb2, ffn2, BS_, E_, F_);

  // residual/LN2 -> output (fp32)
  add_ln_kernel<<<BS_, 256, 0, stream>>>(hf, ffn2, g2, b2, (float*)d_out, nullptr);
}
